// Convolution_68848325755001
// MI455X (gfx1250) — compile-verified, matrix operand order checked by reading, not run on hardware
//
#include <hip/hip_runtime.h>
#include <hip/hip_bf16.h>
#include <stdint.h>

#define N_NODES 8192
#define F_INC   256
#define F_OUTC  256

typedef __attribute__((ext_vector_type(16))) __bf16 v16bf;
typedef __attribute__((ext_vector_type(8)))  float  v8f;
typedef __attribute__((ext_vector_type(4)))  float  f4v;
typedef __attribute__((vector_size(16)))     int    v4i;

union Frag16 { v16bf v; uint4 q[2]; };

#if __has_builtin(__builtin_amdgcn_global_load_async_to_lds_b128) && \
    __has_builtin(__builtin_amdgcn_s_wait_asynccnt)
#define USE_ASYNC_LDS 1
#else
#define USE_ASYNC_LDS 0
#endif

#if USE_ASYNC_LDS
// per-lane 16B copy global->LDS, ASYNCcnt-tracked; imm offset applies to both sides
// builtin signature (from diagnostics): (int4 as1* src, int4 as3* dst, Imm off, Imm cpol)
#define ASYNC_CP16(g, l, off)                                           \
    __builtin_amdgcn_global_load_async_to_lds_b128(                     \
        (__attribute__((address_space(1))) v4i*)(g),                    \
        (__attribute__((address_space(3))) v4i*)(l), (off), 0)
#endif

__device__ __forceinline__ unsigned short f32_to_bf16_bits(float f) {
    unsigned u = __float_as_uint(f);
    u += 0x7FFFu + ((u >> 16) & 1u);   // round-to-nearest-even
    return (unsigned short)(u >> 16);
}
__device__ __forceinline__ unsigned pack2bf(float a, float b) {
    return (unsigned)f32_to_bf16_bits(a) | ((unsigned)f32_to_bf16_bits(b) << 16);
}

// ---------------------------------------------------------------------------
// Kernel 1: degree diag -> rsqrt, init first-neighbor index
// ---------------------------------------------------------------------------
__global__ void __launch_bounds__(256)
prep_deg_kernel(const float* __restrict__ D, float* __restrict__ rsq,
                int* __restrict__ fidx) {
    int i = blockIdx.x * 256 + threadIdx.x;
    if (i < N_NODES) {
        float d = D[(size_t)i * (N_NODES + 1)];
        rsq[i]  = rsqrtf(d);
        fidx[i] = 0x7FFFFFFF;
    }
}

// ---------------------------------------------------------------------------
// Kernel 2: YT[f][j] = bf16( X[j][f] * rsqrt(deg[j]) )   (tiled transpose)
// ---------------------------------------------------------------------------
__global__ void __launch_bounds__(1024)
make_yt_kernel(const float* __restrict__ X, const float* __restrict__ rsq,
               unsigned short* __restrict__ YT) {
    __shared__ float tile[32][33];
    const int tx = threadIdx.x, ty = threadIdx.y;
    const int j0 = blockIdx.x * 32, f0 = blockIdx.y * 32;
    tile[ty][tx] = X[(size_t)(j0 + ty) * F_INC + f0 + tx];
    __syncthreads();
    float s = rsq[j0 + tx];
    YT[(size_t)(f0 + ty) * N_NODES + j0 + tx] = f32_to_bf16_bits(tile[tx][ty] * s);
}

// ---------------------------------------------------------------------------
// Kernel 3: W fp32 -> bf16
// ---------------------------------------------------------------------------
__global__ void __launch_bounds__(256)
w_bf16_kernel(const float* __restrict__ W, unsigned short* __restrict__ Wb) {
    int i = blockIdx.x * 256 + threadIdx.x;
    Wb[i] = f32_to_bf16_bits(W[i]);
}

// ---------------------------------------------------------------------------
// Kernel 4 (GEMM1): agg = A01(bf16) @ Y(bf16), f32 accum, 64x256 tile per WG.
// Software pipeline: A tile register-prefetched (nontemporal), Y^T tile moved
// by async global->LDS into a double buffer (ASYNCcnt), WMMA overlapped.
// ---------------------------------------------------------------------------
__global__ void __launch_bounds__(256)
spmm_kernel(const float* __restrict__ Aadj, const unsigned short* __restrict__ YT,
            float* __restrict__ agg, int* __restrict__ fidx) {
    // padded rows: 40 ushorts = 80 B (16B-aligned b128 fragment halves)
    __shared__ __align__(16) unsigned short Ash[64 * 40];        //  5 KB
    __shared__ __align__(16) unsigned short Ysh[2][256 * 40];    // 40 KB

    const int tid  = threadIdx.x;
    const int lane = tid & 31;
    const int w    = tid >> 5;
    const int l    = lane & 15;
    const int half = lane >> 4;
    const int wave_m = w & 3;
    const int wave_n = w >> 2;
    const int row0 = blockIdx.x * 64;

    const int arow  = tid >> 2;   // 0..63
    const int aquad = tid & 3;    // 0..3 : 8-wide k chunk
    const float* aptr = Aadj + (size_t)(row0 + arow) * N_NODES + aquad * 8;
    const unsigned short* ytsrc = YT + (size_t)tid * N_NODES;   // this thread's Y^T row

    v8f acc[8] = {};

    // ---- preamble: put tile 0 in flight ----
    f4v fa0 = __builtin_nontemporal_load((const f4v*)aptr);
    f4v fa1 = __builtin_nontemporal_load((const f4v*)aptr + 1);
#if USE_ASYNC_LDS
    {
        unsigned short* l0 = &Ysh[0][tid * 40];
        ASYNC_CP16(ytsrc, l0, 0);
        ASYNC_CP16(ytsrc, l0, 16);
        ASYNC_CP16(ytsrc, l0, 32);
        ASYNC_CP16(ytsrc, l0, 48);
    }
#else
    uint4 y0, y1, y2, y3;
    {
        const uint4* s = (const uint4*)ytsrc;
        y0 = s[0]; y1 = s[1]; y2 = s[2]; y3 = s[3];
    }
#endif

    for (int k0 = 0; k0 < N_NODES; k0 += 32) {
        const int ib = (k0 >> 5) & 1;
        __syncthreads();                     // prev compute done reading LDS

        // ---- A: convert staged regs -> bf16 {0,1} in LDS ----
        {
            const unsigned ONE = 0x3F80u;    // bf16 1.0
            uint4 st;
            st.x = (fa0[0] > 0.f ? ONE : 0u) | ((fa0[1] > 0.f ? ONE : 0u) << 16);
            st.y = (fa0[2] > 0.f ? ONE : 0u) | ((fa0[3] > 0.f ? ONE : 0u) << 16);
            st.z = (fa1[0] > 0.f ? ONE : 0u) | ((fa1[1] > 0.f ? ONE : 0u) << 16);
            st.w = (fa1[2] > 0.f ? ONE : 0u) | ((fa1[3] > 0.f ? ONE : 0u) << 16);
            *(uint4*)&Ash[arow * 40 + aquad * 8] = st;
            // first-neighbor tracking (rarely taken: density ~1/256)
            int kb = k0 + aquad * 8;
            int best = 0x7FFFFFFF;
            if      (fa0[0] > 0.f) best = kb;
            else if (fa0[1] > 0.f) best = kb + 1;
            else if (fa0[2] > 0.f) best = kb + 2;
            else if (fa0[3] > 0.f) best = kb + 3;
            else if (fa1[0] > 0.f) best = kb + 4;
            else if (fa1[1] > 0.f) best = kb + 5;
            else if (fa1[2] > 0.f) best = kb + 6;
            else if (fa1[3] > 0.f) best = kb + 7;
            if (best != 0x7FFFFFFF) atomicMin(&fidx[row0 + arow], best);
        }

        const bool more = (k0 + 32) < N_NODES;
#if USE_ASYNC_LDS
        if (more) {
            // launch next Y^T tile into the other buffer, then wait only for
            // the current tile (async loads complete in order): cnt <= 4
            const unsigned short* gn = ytsrc + k0 + 32;
            unsigned short* ln = &Ysh[ib ^ 1][tid * 40];
            ASYNC_CP16(gn, ln, 0);
            ASYNC_CP16(gn, ln, 16);
            ASYNC_CP16(gn, ln, 32);
            ASYNC_CP16(gn, ln, 48);
            __builtin_amdgcn_s_wait_asynccnt(4);
        } else {
            __builtin_amdgcn_s_wait_asynccnt(0);
        }
#else
        {
            uint4* dst = (uint4*)&Ysh[ib][tid * 40];
            dst[0] = y0; dst[1] = y1; dst[2] = y2; dst[3] = y3;
        }
#endif
        // ---- prefetch next A tile into regs (overlaps with WMMA below) ----
        if (more) {
            const f4v* p = (const f4v*)(aptr + k0 + 32);
            fa0 = __builtin_nontemporal_load(p);
            fa1 = __builtin_nontemporal_load(p + 1);
#if !USE_ASYNC_LDS
            const uint4* s = (const uint4*)(ytsrc + k0 + 32);
            y0 = s[0]; y1 = s[1]; y2 = s[2]; y3 = s[3];
#endif
            if (k0 + 64 < N_NODES) __builtin_prefetch(aptr + k0 + 64, 0, 3);
        }
        __syncthreads();

        // ---- compute: 1 A-frag, 8 distinct B-frags, 8 WMMAs ----
        Frag16 a;
        const int ab = (wave_m * 16 + l) * 40 + half * 8;
        a.q[0] = *(const uint4*)&Ash[ab];
        a.q[1] = *(const uint4*)&Ash[ab + 16];
        const unsigned short* ybase = Ysh[ib];
        Frag16 b[8];
#pragma unroll
        for (int t = 0; t < 8; ++t) {
            const int cb = (wave_n * 128 + t * 16 + l) * 40 + half * 8;
            b[t].q[0] = *(const uint4*)&ybase[cb];
            b[t].q[1] = *(const uint4*)&ybase[cb + 16];
        }
#pragma unroll
        for (int t = 0; t < 8; ++t)
            acc[t] = __builtin_amdgcn_wmma_f32_16x16x32_bf16(
                false, a.v, false, b[t].v, (short)0, acc[t], false, false);
    }

    // ---- write agg (f32): 64B-contiguous per 16-lane group ----
#pragma unroll
    for (int t = 0; t < 8; ++t) {
        const int col = wave_n * 128 + t * 16 + l;
#pragma unroll
        for (int v = 0; v < 8; ++v) {
            const int r = row0 + wave_m * 16 + half * 8 + v;
            agg[(size_t)r * F_INC + col] = acc[t][v];
        }
    }
}

// ---------------------------------------------------------------------------
// Kernel 5 (GEMM2): out = leaky_relu( (s0[i]*agg[i,:]) @ W^T + b )
// ---------------------------------------------------------------------------
__global__ void __launch_bounds__(256)
dense_out_kernel(const float* __restrict__ agg, const unsigned short* __restrict__ Wb,
                 const float* __restrict__ bias, const float* __restrict__ rsq,
                 const int* __restrict__ fidx, float* __restrict__ out) {
    const int tid  = threadIdx.x;
    const int lane = tid & 31;
    const int w    = tid >> 5;
    const int l    = lane & 15;
    const int half = lane >> 4;
    const int wave_m = w & 3;
    const int wave_n = w >> 2;
    const int row0 = blockIdx.x * 64;
    const int row  = row0 + wave_m * 16 + l;
    const float s0 = rsq[fidx[row]];

    v8f acc[8] = {};
    const float* ar = agg + (size_t)row * F_INC;
#pragma unroll
    for (int k0 = 0; k0 < F_INC; k0 += 32) {
        Frag16 a;
        float4 x0 = *(const float4*)(ar + k0 + half * 8);
        float4 x1 = *(const float4*)(ar + k0 + half * 8 + 4);
        float4 x2 = *(const float4*)(ar + k0 + 16 + half * 8);
        float4 x3 = *(const float4*)(ar + k0 + 16 + half * 8 + 4);
        a.q[0].x = pack2bf(x0.x * s0, x0.y * s0);
        a.q[0].y = pack2bf(x0.z * s0, x0.w * s0);
        a.q[0].z = pack2bf(x1.x * s0, x1.y * s0);
        a.q[0].w = pack2bf(x1.z * s0, x1.w * s0);
        a.q[1].x = pack2bf(x2.x * s0, x2.y * s0);
        a.q[1].y = pack2bf(x2.z * s0, x2.w * s0);
        a.q[1].z = pack2bf(x3.x * s0, x3.y * s0);
        a.q[1].w = pack2bf(x3.z * s0, x3.w * s0);
        Frag16 b[8];
#pragma unroll
        for (int t = 0; t < 8; ++t) {
            const int col = wave_n * 128 + t * 16 + l;   // output channel
            b[t].q[0] = *(const uint4*)(Wb + (size_t)col * F_INC + k0 + half * 8);
            b[t].q[1] = *(const uint4*)(Wb + (size_t)col * F_INC + k0 + 16 + half * 8);
        }
#pragma unroll
        for (int t = 0; t < 8; ++t)
            acc[t] = __builtin_amdgcn_wmma_f32_16x16x32_bf16(
                false, a.v, false, b[t].v, (short)0, acc[t], false, false);
    }
#pragma unroll
    for (int t = 0; t < 8; ++t) {
        const int col = wave_n * 128 + t * 16 + l;
        const float bv = bias[col];
#pragma unroll
        for (int v = 0; v < 8; ++v) {
            const int r = row0 + wave_m * 16 + half * 8 + v;
            float x = acc[t][v] + bv;
            out[(size_t)r * F_OUTC + col] = (x > 0.f) ? x : 0.01f * x;
        }
    }
}

// ---------------------------------------------------------------------------
extern "C" void kernel_launch(void* const* d_in, const int* in_sizes, int n_in,
                              void* d_out, int out_size, void* d_ws, size_t ws_size,
                              hipStream_t stream) {
    (void)in_sizes; (void)n_in; (void)out_size; (void)ws_size;
    const float* D = (const float*)d_in[0];
    const float* X = (const float*)d_in[1];
    const float* A = (const float*)d_in[2];
    const float* W = (const float*)d_in[3];
    const float* b = (const float*)d_in[4];
    float* out = (float*)d_out;

    char* ws = (char*)d_ws;
    float*          agg  = (float*)(ws);                      //  8 MB  [8192,256] f32
    unsigned short* YT   = (unsigned short*)(ws + 8388608);   //  4 MB  [256,8192] bf16
    unsigned short* Wb   = (unsigned short*)(ws + 12582912);  // 128 KB [256,256]  bf16
    float*          rsq  = (float*)(ws + 12713984);           // 32 KB
    int*            fidx = (int*)(ws + 12746752);             // 32 KB

    prep_deg_kernel <<<N_NODES / 256, 256, 0, stream>>>(D, rsq, fidx);
    make_yt_kernel  <<<dim3(N_NODES / 32, F_INC / 32), dim3(32, 32), 0, stream>>>(X, rsq, YT);
    w_bf16_kernel   <<<(F_OUTC * F_INC) / 256, 256, 0, stream>>>(W, Wb);
    spmm_kernel     <<<N_NODES / 64, 256, 0, stream>>>(A, YT, agg, fidx);
    dense_out_kernel<<<N_NODES / 64, 256, 0, stream>>>(agg, Wb, b, rsq, fidx, out);
}
